// ManualSpecialTransformer_45148696215772
// MI455X (gfx1250) — matrix operand order repaired and checked by reading, not compiled, not run on hardware
//
#include <hip/hip_runtime.h>
#include <math.h>

// ---------------------------------------------------------------------------
// Types for CDNA5 WMMA (wave32, V_WMMA_F32_16X16X32_BF16)
// ---------------------------------------------------------------------------
typedef __bf16 bf16_t;
typedef __attribute__((ext_vector_type(16))) __bf16 bf16x16;
typedef __attribute__((ext_vector_type(8)))  __bf16 bf16x8;
typedef __attribute__((ext_vector_type(8)))  float  floatx8;

#define HH   6
#define BB   8
#define LL   1024
#define DD   768
#define DKK  128
#define DVV  128
#define DFFN 1536
#define MLL  (BB * LL)        /* 8192 rows */
#define NELEM ((size_t)MLL * DD)
#define EPSLN 1e-14f
#define MASKVAL (-1e30f)

// ---------------------------------------------------------------------------
// CDNA5 async global->LDS copies (ASYNCcnt), guarded so the file always builds
// Builtin signature (from hipcc diagnostic): (int4 AS1*, int4 AS3*, Ii, Ii)
// ---------------------------------------------------------------------------
#if __has_builtin(__builtin_amdgcn_global_load_async_to_lds_b128) && \
    __has_builtin(__builtin_amdgcn_s_wait_asynccnt)
#define ASYNC_OK 1
#else
#define ASYNC_OK 0
#endif

#if ASYNC_OK
typedef int intx4 __attribute__((__vector_size__(4 * sizeof(int))));
typedef __attribute__((address_space(1))) intx4 g_intx4;
typedef __attribute__((address_space(3))) intx4 l_intx4;
#endif

// One 16-byte lane-granular global->LDS copy
static __device__ __forceinline__ void copy16_g2l(const bf16_t* g, bf16_t* l) {
#if ASYNC_OK
  __builtin_amdgcn_global_load_async_to_lds_b128(
      (g_intx4*)(uintptr_t)g, (l_intx4*)(uint32_t)(uintptr_t)l, 0, 0);
#else
  *(bf16x8*)l = *(const bf16x8*)g;
#endif
}
static __device__ __forceinline__ void wait_stage_keep6() {
#if ASYNC_OK
  __builtin_amdgcn_s_wait_asynccnt(6);  // previous stage (6 copies) retired
#endif
}
static __device__ __forceinline__ void wait_stage_all() {
#if ASYNC_OK
  __builtin_amdgcn_s_wait_asynccnt(0);
#endif
}

static __device__ __forceinline__ floatx8 zero8() {
  floatx8 z;
#pragma unroll
  for (int i = 0; i < 8; ++i) z[i] = 0.0f;
  return z;
}

// A-operand (16x32 bf16, MxK): lane holds row M=lane&15.
// element e<8  -> K = hi*8 + e ; element e>=8 -> K = 16 + hi*8 + (e-8)
static __device__ __forceinline__ bf16x16 load_a(const bf16_t* rowbase, int koff) {
  bf16x8 lo = *(const bf16x8*)(rowbase + koff);
  bf16x8 hh = *(const bf16x8*)(rowbase + 16 + koff);
  bf16x16 r;
#pragma unroll
  for (int i = 0; i < 8; ++i) { r[i] = lo[i]; r[8 + i] = hh[i]; }
  return r;
}

// B-operand (32x16 bf16, KxN) stored as [N][K] rows: lane holds col N=lane&15.
// element e<8  -> K = hi*8 + e ; element e>=8 -> K = 16 + hi*8 + (e-8)
static __device__ __forceinline__ bf16x16 load_b(const bf16_t* rowbase, int hi) {
  bf16x8 lo = *(const bf16x8*)(rowbase + hi * 8);
  bf16x8 hh = *(const bf16x8*)(rowbase + 16 + hi * 8);
  bf16x16 r;
#pragma unroll
  for (int i = 0; i < 8; ++i) { r[i] = lo[i]; r[8 + i] = hh[i]; }
  return r;
}

static __device__ __forceinline__ floatx8 wmma_bf16(bf16x16 a, bf16x16 b, floatx8 c) {
  // (neg_a, A, neg_b, B, c_mod, C, reuse_a, reuse_b)
  return __builtin_amdgcn_wmma_f32_16x16x32_bf16(false, a, false, b, (short)0, c,
                                                 false, false);
}

// ---------------------------------------------------------------------------
// 1) Elementwise prep: pair sums (bf16) + triple sum (f32 + bf16)
//    pairs[0]=t+v (0,1)  pairs[1]=t+r (0,2)  pairs[2]=v+r (1,2)
// ---------------------------------------------------------------------------
__global__ void prep_acts(const float* __restrict__ t, const float* __restrict__ v,
                          const float* __restrict__ r, bf16_t* __restrict__ pairs,
                          bf16_t* __restrict__ sum3b, float* __restrict__ sum3f) {
  size_t i = (size_t)blockIdx.x * blockDim.x + threadIdx.x;
  if (i >= NELEM) return;
  float a = t[i], b = v[i], c = r[i];
  pairs[0 * NELEM + i] = (bf16_t)(a + b);
  pairs[1 * NELEM + i] = (bf16_t)(a + c);
  pairs[2 * NELEM + i] = (bf16_t)(b + c);
  float s = a + b + c;
  sum3f[i] = s;
  sum3b[i] = (bf16_t)s;
}

// ---------------------------------------------------------------------------
// 2) Weight convert + transpose: W[K][N] f32 -> WT[N][K] bf16 (batched over y)
// ---------------------------------------------------------------------------
__global__ void conv_wt(const float* __restrict__ W, bf16_t* __restrict__ WT,
                        int K, int N) {
  int z = blockIdx.y;
  const float* Wz = W + (size_t)z * K * N;
  bf16_t* Tz = WT + (size_t)z * K * N;
  int i = blockIdx.x * blockDim.x + threadIdx.x;
  if (i >= K * N) return;
  int k = i / N, n = i - k * N;
  Tz[(size_t)n * K + k] = (bf16_t)Wz[i];
}

// ---------------------------------------------------------------------------
// 3) QKV projection GEMM (WMMA, BK=64, double-buffered async LDS staging).
//    grid = (1, M/64, 18). z -> (head, which)
//    q,k stored [H][B*L][DK] bf16; v stored transposed [H][B][DV][L] bf16.
// ---------------------------------------------------------------------------
__global__ __launch_bounds__(256) void qkv_gemm(
    const bf16_t* __restrict__ pairs, const bf16_t* __restrict__ sum3b,
    const bf16_t* __restrict__ WqT, const bf16_t* __restrict__ WkT,
    const bf16_t* __restrict__ WvT, bf16_t* __restrict__ q,
    bf16_t* __restrict__ k, bf16_t* __restrict__ vT) {
  const int QSEL[6] = {1, 2, 1, 0, 2, 1};
  const int KSEL[6] = {1, 2, 2, 0, 0, 0};
  int z = blockIdx.z;
  int h = z / 3, which = z - 3 * h;
  const bf16_t* A;
  const bf16_t* WT;
  if (which == 0)      { A = pairs + (size_t)QSEL[h] * NELEM; WT = WqT + (size_t)h * DKK * DD; }
  else if (which == 1) { A = pairs + (size_t)KSEL[h] * NELEM; WT = WkT + (size_t)h * DKK * DD; }
  else                 { A = sum3b;                           WT = WvT + (size_t)h * DKK * DD; }

  __shared__ __align__(16) bf16_t As[2][64][72];
  __shared__ __align__(16) bf16_t Bs[2][128][72];

  int tid = threadIdx.x, lane = tid & 31, w = tid >> 5;
  int wm = w & 3, wn = w >> 2;
  int nn = lane & 15, hi = lane >> 4, koff = hi * 8;
  int m0 = blockIdx.y * 64;
  int crow = tid >> 3, cseg = (tid & 7) * 8;

  floatx8 acc[4];
#pragma unroll
  for (int i = 0; i < 4; ++i) acc[i] = zero8();

  auto issue = [&](int buf, int k0) {
    const bf16_t* Ag = A + (size_t)(m0 + crow) * DD + k0 + cseg;
    copy16_g2l(Ag,                  &As[buf][crow][cseg]);
    copy16_g2l(Ag + (size_t)32 * DD, &As[buf][32 + crow][cseg]);
    const bf16_t* Bg = WT + (size_t)crow * DD + k0 + cseg;
    copy16_g2l(Bg,                  &Bs[buf][crow][cseg]);
    copy16_g2l(Bg + (size_t)32 * DD, &Bs[buf][32 + crow][cseg]);
    copy16_g2l(Bg + (size_t)64 * DD, &Bs[buf][64 + crow][cseg]);
    copy16_g2l(Bg + (size_t)96 * DD, &Bs[buf][96 + crow][cseg]);
  };

  const int nstage = DD >> 6;  // 12
  issue(0, 0);
  for (int s = 0; s < nstage; ++s) {
    int buf = s & 1;
    if (s + 1 < nstage) {
      __syncthreads();                  // other buffer free for overwrite
      issue(buf ^ 1, (s + 1) << 6);     // prefetch next stage (async DMA)
      wait_stage_keep6();               // this stage's 6 copies retired
    } else {
      wait_stage_all();
    }
    __syncthreads();                    // all waves' copies visible
    bf16x16 a0 = load_a(&As[buf][wm * 16 + nn][0],  koff);
    bf16x16 a1 = load_a(&As[buf][wm * 16 + nn][32], koff);
#pragma unroll
    for (int nt = 0; nt < 4; ++nt) {
      const bf16_t* br = &Bs[buf][wn * 64 + nt * 16 + nn][0];
      acc[nt] = wmma_bf16(a0, load_b(br, hi),      acc[nt]);
      acc[nt] = wmma_bf16(a1, load_b(br + 32, hi), acc[nt]);
    }
  }

  if (which < 2) {
    bf16_t* out = (which == 0 ? q : k) + (size_t)h * MLL * DKK;
#pragma unroll
    for (int nt = 0; nt < 4; ++nt)
#pragma unroll
      for (int i = 0; i < 8; ++i) {
        int m = m0 + wm * 16 + i + 8 * hi;
        int n = wn * 64 + nt * 16 + nn;
        out[(size_t)m * DKK + n] = (bf16_t)acc[nt][i];
      }
  } else {
#pragma unroll
    for (int nt = 0; nt < 4; ++nt)
#pragma unroll
      for (int i = 0; i < 8; ++i) {
        int m = m0 + wm * 16 + i + 8 * hi;
        int dv = wn * 64 + nt * 16 + nn;
        int b = m >> 10, l = m & 1023;
        vT[(((size_t)h * BB + b) * DVV + dv) * LL + l] = (bf16_t)acc[nt][i];
      }
  }
}

// ---------------------------------------------------------------------------
// 4) Flash attention (WMMA). grid = (L/64, B, H), block = 128 (4 waves).
//    Each wave: 16 Q rows x full DV=128. Online softmax via per-wave LDS
//    (same-wave LDS ops are hardware-in-order -> no barriers needed).
// ---------------------------------------------------------------------------
__global__ __launch_bounds__(128) void attention(
    const bf16_t* __restrict__ q, const bf16_t* __restrict__ k,
    const bf16_t* __restrict__ vT, const int* __restrict__ mask,
    bf16_t* __restrict__ conc) {
  __shared__ __align__(16) float  Sbuf[4][16][33];
  __shared__ __align__(16) bf16_t Pbuf[4][16][40];
  __shared__ float scl[4][16];
  __shared__ float inv[4][16];

  int tid = threadIdx.x, lane = tid & 31, w = tid >> 5;
  int nn = lane & 15, hi = lane >> 4, koff = hi * 8;
  int h = blockIdx.z, bb = blockIdx.y;
  int m0 = blockIdx.x * 64 + w * 16;
  size_t hb = (size_t)h * BB + bb;
  const bf16_t* kbase = k + hb * LL * DKK;
  const bf16_t* vbase = vT + hb * DVV * LL;

  // Q tile held in registers as 4 A-operands (DK = 4 x 32)
  const bf16_t* qrow = q + (hb * LL + m0 + nn) * DKK;
  bf16x16 aq[4];
#pragma unroll
  for (int c = 0; c < 4; ++c) aq[c] = load_a(qrow + c * 32, koff);

  floatx8 acc[8];
#pragma unroll
  for (int i = 0; i < 8; ++i) acc[i] = zero8();

  float run_max = -__builtin_inff(), run_sum = 0.0f;
  const int* mrow = mask + (size_t)bb * LL;

  for (int l0 = 0; l0 < LL; l0 += 32) {
    if (l0 + 32 < LL) {  // prefetch next K/V chunk (global_prefetch_b8)
      __builtin_prefetch(kbase + (size_t)(l0 + 32 + nn) * DKK, 0, 3);
      __builtin_prefetch(vbase + (size_t)(nn * 8 + hi) * LL + l0 + 32, 0, 3);
    }
    // ---- S = Q K^T  (16 x 32), keys are B-operand rows [l][dk] -------------
    floatx8 s0 = zero8(), s1 = zero8();
#pragma unroll
    for (int c = 0; c < 4; ++c) {
      bf16x16 b0 = load_b(kbase + (size_t)(l0 + nn) * DKK + c * 32, hi);
      s0 = wmma_bf16(aq[c], b0, s0);
      bf16x16 b1 = load_b(kbase + (size_t)(l0 + 16 + nn) * DKK + c * 32, hi);
      s1 = wmma_bf16(aq[c], b1, s1);
    }
#pragma unroll
    for (int i = 0; i < 8; ++i) {
      Sbuf[w][i + 8 * hi][nn]      = s0[i];
      Sbuf[w][i + 8 * hi][16 + nn] = s1[i];
    }
    // ---- online softmax (lanes 0..15, row r = lane) ------------------------
    if (lane < 16) {
      int r = lane;
      float cm = run_max;
#pragma unroll 4
      for (int j = 0; j < 32; ++j) {
        float sv = Sbuf[w][r][j];
        if (mrow[l0 + j] == 0) sv = MASKVAL;
        Sbuf[w][r][j] = sv;
        cm = fmaxf(cm, sv);
      }
      float corr = __expf(run_max - cm);
      float sum = run_sum * corr;
#pragma unroll 4
      for (int j = 0; j < 32; ++j) {
        float p = __expf(Sbuf[w][r][j] - cm);
        sum += p;
        Pbuf[w][r][j] = (bf16_t)p;
      }
      run_max = cm; run_sum = sum;
      scl[w][r] = corr;
    }
    // ---- rescale accumulators ---------------------------------------------
    float c8[8];
#pragma unroll
    for (int i = 0; i < 8; ++i) c8[i] = scl[w][i + 8 * hi];
#pragma unroll
    for (int nt = 0; nt < 8; ++nt)
#pragma unroll
      for (int i = 0; i < 8; ++i) acc[nt][i] *= c8[i];
    // ---- O += P * V  (V already [dv][l] -> B-operand rows) -----------------
    bf16x16 ap = load_a(&Pbuf[w][nn][0], koff);
#pragma unroll
    for (int nt = 0; nt < 8; ++nt) {
      bf16x16 bv = load_b(vbase + (size_t)(nt * 16 + nn) * LL + l0, hi);
      acc[nt] = wmma_bf16(ap, bv, acc[nt]);
    }
  }

  if (lane < 16) inv[w][lane] = 1.0f / run_sum;
  float i8[8];
#pragma unroll
  for (int i = 0; i < 8; ++i) i8[i] = inv[w][i + 8 * hi];
#pragma unroll
  for (int nt = 0; nt < 8; ++nt)
#pragma unroll
    for (int i = 0; i < 8; ++i) {
      int m = m0 + i + 8 * hi;
      int col = h * DVV + nt * 16 + nn;
      conc[((size_t)bb * LL + m) * (HH * DVV) + col] = (bf16_t)(acc[nt][i] * i8[i]);
    }
}

// ---------------------------------------------------------------------------
// 5) Generic WMMA GEMM (BK=64, double-buffered async LDS staging):
//    C[M x N] = A[M x K] * WT[N x K]^T, M = 8192, K % 64 == 0, N % 128 == 0.
//    MODE 0: f32 out. MODE 2: bias + relu -> bf16 out. MODE 3: bias -> f32 out
// ---------------------------------------------------------------------------
template <int MODE>
__global__ __launch_bounds__(256) void gemm_ws(
    const bf16_t* __restrict__ A, const bf16_t* __restrict__ WT,
    float* __restrict__ Cf, bf16_t* __restrict__ Cb,
    const float* __restrict__ bias, int N, int K) {
  __shared__ __align__(16) bf16_t As[2][64][72];
  __shared__ __align__(16) bf16_t Bs[2][128][72];

  int tid = threadIdx.x, lane = tid & 31, w = tid >> 5;
  int wm = w & 3, wn = w >> 2;
  int nn = lane & 15, hi = lane >> 4, koff = hi * 8;
  int m0 = blockIdx.y * 64, n0 = blockIdx.x * 128;
  int crow = tid >> 3, cseg = (tid & 7) * 8;

  floatx8 acc[4];
#pragma unroll
  for (int i = 0; i < 4; ++i) acc[i] = zero8();

  auto issue = [&](int buf, int k0) {
    const bf16_t* Ag = A + (size_t)(m0 + crow) * K + k0 + cseg;
    copy16_g2l(Ag,                  &As[buf][crow][cseg]);
    copy16_g2l(Ag + (size_t)32 * K, &As[buf][32 + crow][cseg]);
    const bf16_t* Bg = WT + (size_t)(n0 + crow) * K + k0 + cseg;
    copy16_g2l(Bg,                  &Bs[buf][crow][cseg]);
    copy16_g2l(Bg + (size_t)32 * K, &Bs[buf][32 + crow][cseg]);
    copy16_g2l(Bg + (size_t)64 * K, &Bs[buf][64 + crow][cseg]);
    copy16_g2l(Bg + (size_t)96 * K, &Bs[buf][96 + crow][cseg]);
  };

  const int nstage = K >> 6;
  issue(0, 0);
  for (int s = 0; s < nstage; ++s) {
    int buf = s & 1;
    if (s + 1 < nstage) {
      __syncthreads();
      issue(buf ^ 1, (s + 1) << 6);
      wait_stage_keep6();
    } else {
      wait_stage_all();
    }
    __syncthreads();
    bf16x16 a0 = load_a(&As[buf][wm * 16 + nn][0],  koff);
    bf16x16 a1 = load_a(&As[buf][wm * 16 + nn][32], koff);
#pragma unroll
    for (int nt = 0; nt < 4; ++nt) {
      const bf16_t* br = &Bs[buf][wn * 64 + nt * 16 + nn][0];
      acc[nt] = wmma_bf16(a0, load_b(br, hi),      acc[nt]);
      acc[nt] = wmma_bf16(a1, load_b(br + 32, hi), acc[nt]);
    }
  }

#pragma unroll
  for (int nt = 0; nt < 4; ++nt)
#pragma unroll
    for (int i = 0; i < 8; ++i) {
      int m = m0 + wm * 16 + i + 8 * hi;
      int n = n0 + wn * 64 + nt * 16 + nn;
      float v = acc[nt][i];
      if (MODE == 2) {
        v = fmaxf(v + bias[n], 0.0f);
        Cb[(size_t)m * N + n] = (bf16_t)v;
      } else if (MODE == 3) {
        Cf[(size_t)m * N + n] = v + bias[n];
      } else {
        Cf[(size_t)m * N + n] = v;
      }
    }
}

// ---------------------------------------------------------------------------
// 6) x = LayerNorm(X + Y) * gamma[gi] + beta[gi]; optional bf16 copy.
//    One block per row (768 cols, 256 threads x 3).
// ---------------------------------------------------------------------------
__global__ __launch_bounds__(256) void add_ln(
    const float* __restrict__ X, const float* __restrict__ Y,
    const float* __restrict__ gamma, const float* __restrict__ beta, int gi,
    float* __restrict__ outF, bf16_t* __restrict__ outB) {
  int row = blockIdx.x, tid = threadIdx.x;
  const float* x = X + (size_t)row * DD;
  const float* y = Y + (size_t)row * DD;
  float v[3], s = 0.0f, s2 = 0.0f;
#pragma unroll
  for (int j = 0; j < 3; ++j) {
    float t = x[tid + j * 256] + y[tid + j * 256];
    v[j] = t; s += t; s2 += t * t;
  }
  __shared__ float red[256], red2[256];
  red[tid] = s; red2[tid] = s2;
  __syncthreads();
  for (int off = 128; off > 0; off >>= 1) {
    if (tid < off) { red[tid] += red[tid + off]; red2[tid] += red2[tid + off]; }
    __syncthreads();
  }
  float mean = red[0] * (1.0f / DD);
  float var = fmaxf(red2[0] * (1.0f / DD) - mean * mean, 0.0f);
  float rstd = rsqrtf(var + EPSLN);
  float g = gamma[gi], b = beta[gi];
#pragma unroll
  for (int j = 0; j < 3; ++j) {
    float o = (v[j] - mean) * rstd * g + b;
    outF[(size_t)row * DD + tid + j * 256] = o;
    if (outB) outB[(size_t)row * DD + tid + j * 256] = (bf16_t)o;
  }
}

// ---------------------------------------------------------------------------
// Host launcher
// ---------------------------------------------------------------------------
extern "C" void kernel_launch(void* const* d_in, const int* in_sizes, int n_in,
                              void* d_out, int out_size, void* d_ws, size_t ws_size,
                              hipStream_t stream) {
  (void)in_sizes; (void)n_in; (void)out_size; (void)ws_size;
  const float* times  = (const float*)d_in[0];
  const float* values = (const float*)d_in[1];
  const float* varis  = (const float*)d_in[2];
  const int*   mask   = (const int*)d_in[3];
  const float* Wq = (const float*)d_in[4];
  const float* Wk = (const float*)d_in[5];
  const float* Wv = (const float*)d_in[6];
  const float* Wo = (const float*)d_in[7];
  const float* W1 = (const float*)d_in[8];
  const float* b1 = (const float*)d_in[9];
  const float* W2 = (const float*)d_in[10];
  const float* b2 = (const float*)d_in[11];
  const float* gamma = (const float*)d_in[12];
  const float* beta  = (const float*)d_in[13];

  char* p = (char*)d_ws;
  auto alloc = [&](size_t bytes) -> char* {
    char* r = p;
    p += (bytes + 255) & ~(size_t)255;
    return r;
  };
  bf16_t* pairs = (bf16_t*)alloc(3 * NELEM * sizeof(bf16_t));
  bf16_t* sum3b = (bf16_t*)alloc(NELEM * sizeof(bf16_t));
  float*  sum3f = (float*) alloc(NELEM * sizeof(float));
  bf16_t* qb    = (bf16_t*)alloc((size_t)HH * MLL * DKK * sizeof(bf16_t));
  bf16_t* kb    = (bf16_t*)alloc((size_t)HH * MLL * DKK * sizeof(bf16_t));
  bf16_t* vTb   = (bf16_t*)alloc((size_t)HH * BB * DVV * LL * sizeof(bf16_t));
  bf16_t* conc  = (bf16_t*)alloc((size_t)MLL * HH * DVV * sizeof(bf16_t));
  float*  proj  = (float*) alloc(NELEM * sizeof(float));
  float*  x1f   = (float*) alloc(NELEM * sizeof(float));
  bf16_t* x1b   = (bf16_t*)alloc(NELEM * sizeof(bf16_t));
  bf16_t* ffn1  = (bf16_t*)alloc((size_t)MLL * DFFN * sizeof(bf16_t));
  float*  ffn2  = (float*) alloc(NELEM * sizeof(float));
  bf16_t* WqT   = (bf16_t*)alloc((size_t)HH * DD * DKK * sizeof(bf16_t));
  bf16_t* WkT   = (bf16_t*)alloc((size_t)HH * DD * DKK * sizeof(bf16_t));
  bf16_t* WvT   = (bf16_t*)alloc((size_t)HH * DD * DVV * sizeof(bf16_t));
  bf16_t* WoT   = (bf16_t*)alloc((size_t)DD * DD * sizeof(bf16_t));
  bf16_t* W1T   = (bf16_t*)alloc((size_t)DD * DFFN * sizeof(bf16_t));
  bf16_t* W2T   = (bf16_t*)alloc((size_t)DFFN * DD * sizeof(bf16_t));

  // 1) activations
  prep_acts<<<(unsigned)((NELEM + 255) / 256), 256, 0, stream>>>(
      times, values, varis, pairs, sum3b, sum3f);

  // 2) weights -> bf16 [N][K]
  conv_wt<<<dim3((DD * DKK + 255) / 256, HH), 256, 0, stream>>>(Wq, WqT, DD, DKK);
  conv_wt<<<dim3((DD * DKK + 255) / 256, HH), 256, 0, stream>>>(Wk, WkT, DD, DKK);
  conv_wt<<<dim3((DD * DVV + 255) / 256, HH), 256, 0, stream>>>(Wv, WvT, DD, DVV);
  conv_wt<<<dim3((DD * DD + 255) / 256, 1), 256, 0, stream>>>(Wo, WoT, DD, DD);
  conv_wt<<<dim3((DD * DFFN + 255) / 256, 1), 256, 0, stream>>>(W1, W1T, DD, DFFN);
  conv_wt<<<dim3((DFFN * DD + 255) / 256, 1), 256, 0, stream>>>(W2, W2T, DFFN, DD);

  // 3) QKV projections (18 GEMMs fused into one launch)
  qkv_gemm<<<dim3(1, MLL / 64, HH * 3), 256, 0, stream>>>(
      pairs, sum3b, WqT, WkT, WvT, qb, kb, vTb);

  // 4) attention -> conc [B*L, H*DV] bf16
  attention<<<dim3(LL / 64, BB, HH), 128, 0, stream>>>(qb, kb, vTb, mask, conc);

  // 5) output projection: proj = conc @ Wo
  gemm_ws<0><<<dim3(DD / 128, MLL / 64), 256, 0, stream>>>(
      conc, WoT, proj, nullptr, nullptr, DD, DD);

  // 6) x1 = LN(value + proj)
  add_ln<<<MLL, 256, 0, stream>>>(sum3f, proj, gamma, beta, 0, x1f, x1b);

  // 7) ffn1 = relu(x1 @ W1 + b1)
  gemm_ws<2><<<dim3(DFFN / 128, MLL / 64), 256, 0, stream>>>(
      x1b, W1T, nullptr, ffn1, b1, DFFN, DD);

  // 8) ffn2 = ffn1 @ W2 + b2
  gemm_ws<3><<<dim3(DD / 128, MLL / 64), 256, 0, stream>>>(
      ffn1, W2T, ffn2, nullptr, b2, DD, DFFN);

  // 9) out = LN(x1 + ffn2)
  add_ln<<<MLL, 256, 0, stream>>>(x1f, ffn2, gamma, beta, 1, (float*)d_out, nullptr);
}
